// TransformerLayer_2688649527872
// MI455X (gfx1250) — matrix operand order repaired and checked by reading, not compiled, not run on hardware
//
#include <hip/hip_runtime.h>

typedef __attribute__((ext_vector_type(16))) __bf16 v16bf;
typedef __attribute__((ext_vector_type(8)))  float  v8f;

#define HID   1024
#define SEQ   2048
#define BATCH 4
#define NHEAD 16
#define HDIM  64
#define FFDIM 4096
#define MROWS (BATCH*SEQ)   /* 8192 */

#if defined(__has_builtin)
#  if __has_builtin(__builtin_amdgcn_tensor_load_to_lds)
#    define HAVE_TDM 1
#  endif
#endif
#ifndef HAVE_TDM
#  define HAVE_TDM 0
#endif

// ---------------------------------------------------------------- WMMA core
__device__ inline v8f wmma_bf16(v16bf a, v16bf b, v8f c) {
  return __builtin_amdgcn_wmma_f32_16x16x32_bf16(false, a, false, b, (short)0, c,
                                                 false, false);
}

// ---------------------------------------------------------------- async copy (fallback path)
__device__ inline void async_b128(uint32_t lds_off, uint32_t goff, const void* base) {
  asm volatile("global_load_async_to_lds_b128 %0, %1, %2"
               :: "v"(lds_off), "v"(goff), "s"(base) : "memory");
}
__device__ inline void wait_async0() {
#if __has_builtin(__builtin_amdgcn_s_wait_asynccnt)
  __builtin_amdgcn_s_wait_asynccnt(0);
#else
  asm volatile("s_wait_asynccnt 0" ::: "memory");
#endif
}

// ---------------------------------------------------------------- TDM tile load
#if HAVE_TDM
typedef __attribute__((ext_vector_type(4))) unsigned int u32x4;
typedef __attribute__((ext_vector_type(4))) int i32x4;
typedef __attribute__((ext_vector_type(8))) int i32x8;

// Load a 128-row x 32-element bf16 tile (row stride kdim elements) into LDS,
// hardware-padding 16B after every 64B row -> LDS row stride 40 bf16.
__device__ inline void tdm_load_tile(uint32_t lds_addr, const __bf16* gptr,
                                     uint32_t kdim, uint32_t rows) {
  const uint64_t ga = (uint64_t)(uintptr_t)gptr;
  const u32x4 g0 = { 1u,                                   // count=1 (valid D#)
                     lds_addr,                             // lds_addr
                     (uint32_t)ga,                         // global_addr[31:0]
                     (uint32_t)((ga >> 32) & 0x01ffffffu)  // global_addr[56:32]
                       | 0x80000000u };                    // type=2 ("image")
  const i32x8 g1 = {
      (int)((1u << 16)        // data_size = 2 bytes
            | (1u << 20)      // pad_enable
            | (3u << 22)      // pad_interval: every 16 DWORDs (64B)
            | (3u << 25)),    // pad_amount: 4 DWORDs (16B)
      (int)((kdim & 0xffffu) << 16),                                   // dim0[15:0]
      (int)(((kdim >> 16) & 0xffffu) | ((rows & 0xffffu) << 16)),      // dim0[31:16] | dim1[15:0]
      (int)(((rows >> 16) & 0xffffu) | (32u << 16)),                   // dim1[31:16] | tile_dim0=32
      (int)128,                                                        // tile_dim1=128, tile_dim2=0
      (int)kdim,                                                       // dim0_stride[31:0]
      0, 0 };
  const i32x4 g2 = {0, 0, 0, 0};
  const i32x4 g3 = {0, 0, 0, 0};
#if defined(__clang_major__) && (__clang_major__ >= 23)
  const i32x8 g4 = {0, 0, 0, 0, 0, 0, 0, 0};
  __builtin_amdgcn_tensor_load_to_lds(g0, g1, g2, g3, g4, 0);
#else
  __builtin_amdgcn_tensor_load_to_lds(g0, g1, g2, g3, 0);
#endif
}
#endif

// ---------------------------------------------------------------- fragments
// LDS tiles use padded row stride of 40 bf16 (80 B: multiple of 16 B, conflict-free).
__device__ inline v16bf ldsFragA(const __bf16* base) {
  const int lane = threadIdx.x & 31;
  const int m = lane & 15, kh = lane >> 4;
  const __bf16* row = base + m * 40;
  v16bf a;
#pragma unroll
  for (int e = 0; e < 8; ++e) { a[e] = row[kh * 8 + e]; a[e + 8] = row[16 + kh * 8 + e]; }
  return a;
}
__device__ inline v16bf ldsFragB(const __bf16* base) {
  const int lane = threadIdx.x & 31;
  const int n = lane & 15, kh = lane >> 4;
  const __bf16* row = base + n * 40;
  v16bf b;
#pragma unroll
  for (int e = 0; e < 8; ++e) { b[e] = row[kh * 8 + e]; b[e + 8] = row[16 + kh * 8 + e]; }
  return b;
}
__device__ inline v16bf rowFragA16(const __bf16* row, int koff) {
  const int kh = (threadIdx.x >> 4) & 1;
  const __bf16* p0 = row + koff + kh * 8;
  const __bf16* p1 = row + koff + 16 + kh * 8;
  v16bf a;
#pragma unroll
  for (int e = 0; e < 8; ++e) { a[e] = p0[e]; a[e + 8] = p1[e]; }
  return a;
}
__device__ inline v16bf rowFragB16(const __bf16* row, int koff) {
  const int kh = (threadIdx.x >> 4) & 1;
  const __bf16* p0 = row + koff + kh * 8;
  const __bf16* p1 = row + koff + 16 + kh * 8;
  v16bf b;
#pragma unroll
  for (int e = 0; e < 8; ++e) { b[e] = p0[e]; b[e + 8] = p1[e]; }
  return b;
}

// ---------------------------------------------------------------- pre-conversion
__global__ __launch_bounds__(256) void cvt_f32_bf16(const float* __restrict__ in,
                                                    __bf16* __restrict__ out) {
  const size_t i = ((size_t)blockIdx.x * 256 + threadIdx.x) * 4;
  const float4 v = *reinterpret_cast<const float4*>(in + i);
  out[i + 0] = (__bf16)v.x;
  out[i + 1] = (__bf16)v.y;
  out[i + 2] = (__bf16)v.z;
  out[i + 3] = (__bf16)v.w;
}

// W[K][N] fp32  ->  Wt[N][K] bf16  (64x64 tiles through LDS)
__global__ __launch_bounds__(256) void transpose_cvt_bf16(
    const float* __restrict__ W, __bf16* __restrict__ Wt, int K, int N) {
  __shared__ float tile[64][65];
  const int k0 = blockIdx.x * 64, n0 = blockIdx.y * 64;
  const int tx = threadIdx.x & 63, ty = threadIdx.x >> 6;
#pragma unroll
  for (int i = 0; i < 16; ++i) {
    const int r = ty + i * 4;
    tile[r][tx] = W[(size_t)(k0 + r) * N + n0 + tx];
  }
  __syncthreads();
#pragma unroll
  for (int i = 0; i < 16; ++i) {
    const int r = ty + i * 4;
    Wt[(size_t)(n0 + r) * K + k0 + tx] = (__bf16)tile[tx][r];
  }
}

// ---------------------------------------------------------------- GEMM
// out = epilogue(A[M,K](bf16) @ Bt[N,K](bf16)^T + bias)
// EPI 0: bias, scatter bf16 to [B,NH,S,HD]      (outB)
// EPI 1: bias + residual(fp32), fp32 out        (outF)
// EPI 2: bias + exact GELU, bf16 out            (outB)
// Tile: 128x128 per WG (8 waves, 32x64 each), K-step 32.
// Staging: TDM descriptors (wave 0 issues, TENSORcnt), double-buffered LDS.
template <int EPI>
__global__ __launch_bounds__(256) void gemm_bf16_wmma(
    const __bf16* __restrict__ A, const __bf16* __restrict__ Bt,
    const float* __restrict__ bias, const float* __restrict__ res,
    float* __restrict__ outF, __bf16* __restrict__ outB, int M, int N, int K) {
  __shared__ __bf16 Asl[2][128][40];
  __shared__ __bf16 Bsl[2][128][40];

  const int tid  = threadIdx.x;
  const int lane = tid & 31;
  const int w    = tid >> 5;
  const int wr   = (w >> 1) * 32;       // wave row offset (M)
  const int wc   = (w & 1) * 64;        // wave col offset (N)
  const int m0   = blockIdx.x * 128;
  const int n0   = blockIdx.y * 128;

  v8f acc[2][4] = {};

#if HAVE_TDM
  const bool issuer = (w == 0);
  auto issue = [&](int kk, int buf) {
    if (issuer) {
      tdm_load_tile((uint32_t)(uintptr_t)&Asl[buf][0][0], A + (size_t)m0 * K + kk,
                    (uint32_t)K, (uint32_t)M);
      tdm_load_tile((uint32_t)(uintptr_t)&Bsl[buf][0][0], Bt + (size_t)n0 * K + kk,
                    (uint32_t)K, (uint32_t)N);
    }
  };
  auto wait_stage = [&]() {
    if (issuer) __builtin_amdgcn_s_wait_tensorcnt((short)0);
  };
#else
  const int c0   = tid * 2;             // 512 16-byte chunks per tile, 2/thread
  const int row0 = c0 >> 2, cw0 = c0 & 3;
  const int row1 = (c0 + 1) >> 2, cw1 = (c0 + 1) & 3;
  auto issue = [&](int kk, int buf) {
    __bf16 (*la)[40] = Asl[buf];
    __bf16 (*lb)[40] = Bsl[buf];
    async_b128((uint32_t)(uintptr_t)(&la[row0][cw0 * 8]),
               (uint32_t)((((size_t)(m0 + row0)) * K + kk) * 2 + cw0 * 16), A);
    async_b128((uint32_t)(uintptr_t)(&la[row1][cw1 * 8]),
               (uint32_t)((((size_t)(m0 + row1)) * K + kk) * 2 + cw1 * 16), A);
    async_b128((uint32_t)(uintptr_t)(&lb[row0][cw0 * 8]),
               (uint32_t)((((size_t)(n0 + row0)) * K + kk) * 2 + cw0 * 16), Bt);
    async_b128((uint32_t)(uintptr_t)(&lb[row1][cw1 * 8]),
               (uint32_t)((((size_t)(n0 + row1)) * K + kk) * 2 + cw1 * 16), Bt);
  };
  auto wait_stage = [&]() { wait_async0(); };
#endif

  issue(0, 0);
  const int nk = K / 32;
  for (int kt = 0; kt < nk; ++kt) {
    const int cur = kt & 1;
    wait_stage();        // tile kt has landed (issuing wave)
    __syncthreads();     // tile kt visible to all; tile kt-1 reads finished
    if (kt + 1 < nk) issue((kt + 1) * 32, 1 - cur);

    const v16bf a0 = ldsFragA(&Asl[cur][wr][0]);
    const v16bf a1 = ldsFragA(&Asl[cur][wr + 16][0]);
    v16bf b[4];
#pragma unroll
    for (int j = 0; j < 4; ++j) b[j] = ldsFragB(&Bsl[cur][wc + 16 * j][0]);
#pragma unroll
    for (int j = 0; j < 4; ++j) {
      acc[0][j] = wmma_bf16(a0, b[j], acc[0][j]);
      acc[1][j] = wmma_bf16(a1, b[j], acc[1][j]);
    }
  }

  const int nl = lane & 15;
  const int mh = (lane >> 4) * 8;
#pragma unroll
  for (int i = 0; i < 2; ++i)
#pragma unroll
    for (int j = 0; j < 4; ++j) {
      const int n = n0 + wc + j * 16 + nl;
      const float bv = bias[n];
#pragma unroll
      for (int r = 0; r < 8; ++r) {
        const int m = m0 + wr + i * 16 + mh + r;
        float v = acc[i][j][r] + bv;
        if constexpr (EPI == 0) {
          const int bb = m >> 11, s = m & (SEQ - 1);
          const int h = n >> 6, hd = n & (HDIM - 1);
          outB[(((size_t)(bb * NHEAD + h) * SEQ) + s) * HDIM + hd] = (__bf16)v;
        } else if constexpr (EPI == 1) {
          v += res[(size_t)m * N + n];
          outF[(size_t)m * N + n] = v;
        } else {  // GELU -> bf16
          v = 0.5f * v * (1.0f + erff(v * 0.70710678118654752f));
          outB[(size_t)m * N + n] = (__bf16)v;
        }
      }
    }
}

// ---------------------------------------------------------------- attention
// grid: (SEQ/64, BATCH*NHEAD), 128 threads (4 waves, 16 query rows each).
__global__ __launch_bounds__(128) void flash_attn_wmma(
    const __bf16* __restrict__ Q, const __bf16* __restrict__ Kt,
    const __bf16* __restrict__ V, const float* __restrict__ mask,
    __bf16* __restrict__ ctxout) {
  __shared__ __bf16 Vt[64][40];        // V block transposed: [hd][key]
  __shared__ __bf16 Pbuf[4][16][40];   // per-wave P relayout buffer

  const int bh    = blockIdx.y;
  const int b     = bh >> 4;
  const int h     = bh & 15;
  const int lane  = threadIdx.x & 31;
  const int w     = threadIdx.x >> 5;
  const int nloc  = lane & 15;
  const int mbase = (lane >> 4) * 8;
  const float* maskrow = mask + (size_t)b * SEQ;

  const int qrow = blockIdx.x * 64 + w * 16 + nloc;
  const __bf16* qp = Q + ((size_t)bh * SEQ + qrow) * HDIM;
  const v16bf qa0 = rowFragA16(qp, 0);
  const v16bf qa1 = rowFragA16(qp, 32);

  v8f ctx[4] = {};
  float mrow[8], lrow[8];
#pragma unroll
  for (int r = 0; r < 8; ++r) { mrow[r] = -1e30f; lrow[r] = 0.0f; }

  for (int kb = 0; kb < SEQ; kb += 32) {
    __syncthreads();
    {  // stage V block (32 keys x 64 hd) transposed into LDS
      const int key = threadIdx.x >> 2;
      const int hdb = (threadIdx.x & 3) * 16;
      const __bf16* vp = V + ((size_t)bh * SEQ + kb + key) * HDIM + hdb;
#pragma unroll
      for (int e = 0; e < 16; ++e) Vt[hdb + e][key] = vp[e];
    }
    __syncthreads();

    v8f st[2];
#pragma unroll
    for (int t = 0; t < 2; ++t) {
      const __bf16* kp = Kt + ((size_t)bh * SEQ + kb + t * 16 + nloc) * HDIM;
      const v16bf bk0 = rowFragB16(kp, 0);
      const v16bf bk1 = rowFragB16(kp, 32);
      v8f z = {};
      z = wmma_bf16(qa0, bk0, z);
      z = wmma_bf16(qa1, bk1, z);
      const float mk = maskrow[kb + t * 16 + nloc] * -1e9f;
#pragma unroll
      for (int r = 0; r < 8; ++r) z[r] = z[r] * 0.125f + mk;
      st[t] = z;
    }

    float pr0[8], pr1[8];
#pragma unroll
    for (int r = 0; r < 8; ++r) {
      float v = fmaxf(st[0][r], st[1][r]);
      v = fmaxf(v, __shfl_xor(v, 1, 16));
      v = fmaxf(v, __shfl_xor(v, 2, 16));
      v = fmaxf(v, __shfl_xor(v, 4, 16));
      v = fmaxf(v, __shfl_xor(v, 8, 16));
      const float mnew = fmaxf(mrow[r], v);
      const float sc = __expf(mrow[r] - mnew);
      mrow[r] = mnew;
      pr0[r] = __expf(st[0][r] - mnew);
      pr1[r] = __expf(st[1][r] - mnew);
      float rs = pr0[r] + pr1[r];
      rs += __shfl_xor(rs, 1, 16);
      rs += __shfl_xor(rs, 2, 16);
      rs += __shfl_xor(rs, 4, 16);
      rs += __shfl_xor(rs, 8, 16);
      lrow[r] = lrow[r] * sc + rs;
#pragma unroll
      for (int c = 0; c < 4; ++c) ctx[c][r] *= sc;
    }

#pragma unroll
    for (int r = 0; r < 8; ++r) {
      Pbuf[w][mbase + r][nloc]      = (__bf16)pr0[r];
      Pbuf[w][mbase + r][16 + nloc] = (__bf16)pr1[r];
    }
    const v16bf pa = ldsFragA(&Pbuf[w][0][0]);
#pragma unroll
    for (int c = 0; c < 4; ++c) {
      const v16bf bv = ldsFragB(&Vt[c * 16][0]);
      ctx[c] = wmma_bf16(pa, bv, ctx[c]);
    }
  }

#pragma unroll
  for (int c = 0; c < 4; ++c)
#pragma unroll
    for (int r = 0; r < 8; ++r) {
      const int srow = blockIdx.x * 64 + w * 16 + mbase + r;
      const float val = ctx[c][r] / lrow[r];
      ctxout[((size_t)(b * SEQ + srow)) * HID + h * HDIM + c * 16 + nloc] = (__bf16)val;
    }
}

// ---------------------------------------------------------------- layernorm
template <bool WB>
__global__ __launch_bounds__(256) void layernorm_row(
    const float* __restrict__ x, const float* __restrict__ g,
    const float* __restrict__ beta, float* __restrict__ outF,
    __bf16* __restrict__ outB) {
  __shared__ float red[256];
  const int tid = threadIdx.x;
  const size_t row = blockIdx.x;
  const float4 v = *reinterpret_cast<const float4*>(x + row * HID + tid * 4);

  red[tid] = v.x + v.y + v.z + v.w;
  __syncthreads();
  for (int s = 128; s > 0; s >>= 1) {
    if (tid < s) red[tid] += red[tid + s];
    __syncthreads();
  }
  const float mu = red[0] * (1.0f / HID);
  __syncthreads();

  const float d0 = v.x - mu, d1 = v.y - mu, d2 = v.z - mu, d3 = v.w - mu;
  red[tid] = d0 * d0 + d1 * d1 + d2 * d2 + d3 * d3;
  __syncthreads();
  for (int s = 128; s > 0; s >>= 1) {
    if (tid < s) red[tid] += red[tid + s];
    __syncthreads();
  }
  const float inv = rsqrtf(red[0] * (1.0f / HID) + 1e-12f);

  const int c = tid * 4;
  float4 o;
  o.x = d0 * inv * g[c + 0] + beta[c + 0];
  o.y = d1 * inv * g[c + 1] + beta[c + 1];
  o.z = d2 * inv * g[c + 2] + beta[c + 2];
  o.w = d3 * inv * g[c + 3] + beta[c + 3];
  *reinterpret_cast<float4*>(outF + row * HID + c) = o;
  if constexpr (WB) {
    outB[row * HID + c + 0] = (__bf16)o.x;
    outB[row * HID + c + 1] = (__bf16)o.y;
    outB[row * HID + c + 2] = (__bf16)o.z;
    outB[row * HID + c + 3] = (__bf16)o.w;
  }
}

// ---------------------------------------------------------------- launch
extern "C" void kernel_launch(void* const* d_in, const int* in_sizes, int n_in,
                              void* d_out, int out_size, void* d_ws, size_t ws_size,
                              hipStream_t stream) {
  const float* hidden = (const float*)d_in[0];
  const float* amask  = (const float*)d_in[1];
  const float* Wq = (const float*)d_in[2];  const float* bq = (const float*)d_in[3];
  const float* Wk = (const float*)d_in[4];  const float* bk = (const float*)d_in[5];
  const float* Wv = (const float*)d_in[6];  const float* bv = (const float*)d_in[7];
  const float* Wo = (const float*)d_in[8];  const float* bo = (const float*)d_in[9];
  const float* ln1g = (const float*)d_in[10]; const float* ln1b = (const float*)d_in[11];
  const float* W1 = (const float*)d_in[12]; const float* b1 = (const float*)d_in[13];
  const float* W2 = (const float*)d_in[14]; const float* b2 = (const float*)d_in[15];
  const float* ln2g = (const float*)d_in[16]; const float* ln2b = (const float*)d_in[17];

  const size_t MH = (size_t)MROWS * HID;   // 8388608
  char* p = (char*)d_ws;
  auto take = [&](size_t bytes) { char* r = p; p += (bytes + 255) & ~(size_t)255; return r; };

  __bf16* hiddenB = (__bf16*)take(MH * 2);
  __bf16* QbB = (__bf16*)take(MH * 2);
  __bf16* KbB = (__bf16*)take(MH * 2);
  __bf16* VbB = (__bf16*)take(MH * 2);
  __bf16* ctxB = (__bf16*)take(MH * 2);
  __bf16* xlnB = (__bf16*)take(MH * 2);
  __bf16* ffB  = (__bf16*)take(4 * MH * 2);
  float* y1  = (float*)take(MH * 4);
  float* xln = (float*)take(MH * 4);
  float* y2  = (float*)take(MH * 4);
  __bf16* WqT = (__bf16*)take((size_t)HID * HID * 2);
  __bf16* WkT = (__bf16*)take((size_t)HID * HID * 2);
  __bf16* WvT = (__bf16*)take((size_t)HID * HID * 2);
  __bf16* WoT = (__bf16*)take((size_t)HID * HID * 2);
  __bf16* W1T = (__bf16*)take((size_t)HID * FFDIM * 2);
  __bf16* W2T = (__bf16*)take((size_t)HID * FFDIM * 2);

  cvt_f32_bf16<<<MH / 1024, 256, 0, stream>>>(hidden, hiddenB);
  transpose_cvt_bf16<<<dim3(HID / 64, HID / 64), 256, 0, stream>>>(Wq, WqT, HID, HID);
  transpose_cvt_bf16<<<dim3(HID / 64, HID / 64), 256, 0, stream>>>(Wk, WkT, HID, HID);
  transpose_cvt_bf16<<<dim3(HID / 64, HID / 64), 256, 0, stream>>>(Wv, WvT, HID, HID);
  transpose_cvt_bf16<<<dim3(HID / 64, HID / 64), 256, 0, stream>>>(Wo, WoT, HID, HID);
  transpose_cvt_bf16<<<dim3(HID / 64, FFDIM / 64), 256, 0, stream>>>(W1, W1T, HID, FFDIM);
  transpose_cvt_bf16<<<dim3(FFDIM / 64, HID / 64), 256, 0, stream>>>(W2, W2T, FFDIM, HID);

  const dim3 gH(MROWS / 128, HID / 128);    // 64 x 8
  const dim3 gF(MROWS / 128, FFDIM / 128);  // 64 x 32

  gemm_bf16_wmma<0><<<gH, 256, 0, stream>>>(hiddenB, WqT, bq, nullptr, nullptr, QbB, MROWS, HID, HID);
  gemm_bf16_wmma<0><<<gH, 256, 0, stream>>>(hiddenB, WkT, bk, nullptr, nullptr, KbB, MROWS, HID, HID);
  gemm_bf16_wmma<0><<<gH, 256, 0, stream>>>(hiddenB, WvT, bv, nullptr, nullptr, VbB, MROWS, HID, HID);

  flash_attn_wmma<<<dim3(SEQ / 64, BATCH * NHEAD), 128, 0, stream>>>(QbB, KbB, VbB, amask, ctxB);

  gemm_bf16_wmma<1><<<gH, 256, 0, stream>>>(ctxB, WoT, bo, hidden, y1, nullptr, MROWS, HID, HID);
  layernorm_row<true><<<MROWS, 256, 0, stream>>>(y1, ln1g, ln1b, xln, xlnB);

  gemm_bf16_wmma<2><<<gF, 256, 0, stream>>>(xlnB, W1T, b1, nullptr, nullptr, ffB, MROWS, FFDIM, HID);
  gemm_bf16_wmma<1><<<gH, 256, 0, stream>>>(ffB, W2T, b2, xln, y2, nullptr, MROWS, HID, FFDIM);
  layernorm_row<false><<<MROWS, 256, 0, stream>>>(y2, ln2g, ln2b, (float*)d_out, nullptr);
}
